// MambaBlock_10419590660407
// MI455X (gfx1250) — matrix operand order repaired and checked by reading, not compiled, not run on hardware
//
#include <hip/hip_runtime.h>
#include <hip/hip_bf16.h>
#include <math.h>

// ---------------- problem constants ----------------
#define D_MODEL 1024
#define D_STATE 64
#define D_CONV  4
#define D_INNER 2048
#define BSZ     4
#define TLEN    2048
#define ROWS    (BSZ * TLEN)        // 8192

typedef __bf16 bf16_t;
typedef __attribute__((ext_vector_type(16))) __bf16 v16bf;
typedef __attribute__((ext_vector_type(8)))  __bf16 v8bf;
typedef __attribute__((ext_vector_type(8)))  float  v8f;

__device__ __forceinline__ float sigm(float x) { return 1.0f / (1.0f + __expf(-x)); }

// ---------------- small utility kernels ----------------
__global__ void zero_f32_kernel(float* __restrict__ p, int n) {
    int i = blockIdx.x * blockDim.x + threadIdx.x;
    if (i < n) p[i] = 0.0f;
}

// f32 [K x N] row-major  ->  bf16 [N x K] row-major (weight transpose, once/launch)
__global__ void cvt_transpose_kernel(const float* __restrict__ src,
                                     bf16_t* __restrict__ dst, int K, int N) {
    int i = blockIdx.x * blockDim.x + threadIdx.x;
    if (i >= K * N) return;
    int k = i / N, n = i - k * N;           // coalesced read along n
    dst[(size_t)n * K + k] = (bf16_t)src[i];
}

// ---------------- LayerNorm (one block per row of 1024) ----------------
__global__ __launch_bounds__(256)
void ln_kernel(const float* __restrict__ x, const float* __restrict__ g,
               const float* __restrict__ b, bf16_t* __restrict__ xn) {
    const int row = blockIdx.x;
    const int tid = threadIdx.x;
    const float* xr = x + (size_t)row * D_MODEL;

    float vals[4];
    float s = 0.f, ss = 0.f;
#pragma unroll
    for (int j = 0; j < 4; ++j) {
        float v = xr[tid + 256 * j];
        vals[j] = v; s += v; ss += v * v;
    }
#pragma unroll
    for (int o = 16; o > 0; o >>= 1) { s += __shfl_xor(s, o); ss += __shfl_xor(ss, o); }

    __shared__ float rs[8], rss[8];
    __shared__ float mu_s, rstd_s;
    const int wave = tid >> 5, lane = tid & 31;
    if (lane == 0) { rs[wave] = s; rss[wave] = ss; }
    __syncthreads();
    if (tid == 0) {
        float t1 = 0.f, t2 = 0.f;
        for (int w = 0; w < 8; ++w) { t1 += rs[w]; t2 += rss[w]; }
        float mu = t1 * (1.0f / D_MODEL);
        float var = t2 * (1.0f / D_MODEL) - mu * mu;
        mu_s = mu; rstd_s = rsqrtf(var + 1e-5f);
    }
    __syncthreads();
    const float mu = mu_s, rstd = rstd_s;
#pragma unroll
    for (int j = 0; j < 4; ++j) {
        int c = tid + 256 * j;
        float v = (vals[j] - mu) * rstd * g[c] + b[c];
        xn[(size_t)row * D_MODEL + c] = (bf16_t)v;
    }
}

// ---------------- bf16 WMMA GEMM, 64x128 block tile, K-step 32 -------------
// B is pre-transposed: Bt is [N x K] row-major, so staging and lane fragments
// are all contiguous 16B vector ops (no scalar LDS transpose).
// MODE 0: store bf16            (C -> outBf, ldc = N)
// MODE 1: store f32             (C -> outF,  ldc = N)
// MODE 2: softplus(C + bias[n]) row-reduced, atomicAdd into dsum[row]
// MODE 3: store f32 C + resid   (C + resid -> outF, ldc = N)
template <int MODE>
__global__ __launch_bounds__(256)
void gemm_bf16_kernel(const bf16_t* __restrict__ A, const bf16_t* __restrict__ Bt,
                      int M, int N, int K,
                      bf16_t* __restrict__ outBf, float* __restrict__ outF,
                      const float* __restrict__ bias, const float* __restrict__ resid,
                      float* __restrict__ dsum) {
    constexpr int BM = 64, BN = 128, BK = 32, LDA = BK + 8, LDB = BK + 8;
    __shared__ __align__(16) bf16_t sA[BM * LDA];   // [m][k]
    __shared__ __align__(16) bf16_t sB[BN * LDB];   // [n][k] (already transposed)

    const int tid  = threadIdx.x;
    const int wave = tid >> 5, lane = tid & 31;
    const int wr = wave >> 2, wc = wave & 3;        // 2x4 wave grid of 32x32 tiles
    const int lrow = lane & 15, lhi = lane >> 4;
    const int bM = blockIdx.y * BM, bN = blockIdx.x * BN;

    v8f acc[2][2];
#pragma unroll
    for (int i = 0; i < 2; ++i)
#pragma unroll
        for (int j = 0; j < 2; ++j)
#pragma unroll
            for (int e = 0; e < 8; ++e) acc[i][j][e] = 0.0f;

    const int ar = tid >> 2, ac = (tid & 3) * 8;    // A stage: 64x32, 8 elems/thr
    const int br = tid >> 1, bc8 = (tid & 1) * 16;  // B stage: 128x32, 16 elems/thr

    const bf16_t* aGlob = A  + (size_t)(bM + ar) * K + ac;
    const bf16_t* bGlob = Bt + (size_t)(bN + br) * K + bc8;

    // software pipeline: global -> regs -> (barrier) -> LDS -> compute
    v8bf aR  = *(const v8bf*)(aGlob);
    v8bf bR0 = *(const v8bf*)(bGlob);
    v8bf bR1 = *(const v8bf*)(bGlob + 8);

    for (int k0 = 0; k0 < K; k0 += BK) {
        *(v8bf*)&sA[ar * LDA + ac]      = aR;
        *(v8bf*)&sB[br * LDB + bc8]     = bR0;
        *(v8bf*)&sB[br * LDB + bc8 + 8] = bR1;
        __syncthreads();

        if (k0 + BK < K) {                          // fetch next tile into regs
            aR  = *(const v8bf*)(aGlob + k0 + BK);
            bR0 = *(const v8bf*)(bGlob + k0 + BK);
            bR1 = *(const v8bf*)(bGlob + k0 + BK + 8);
        }
        if (k0 + 2 * BK < K) {                      // prefetch tile after that
            __builtin_prefetch(aGlob + k0 + 2 * BK, 0, 3);
            __builtin_prefetch(bGlob + k0 + 2 * BK, 0, 3);
        }

        v16bf af[2], bfr[2];
#pragma unroll
        for (int mi = 0; mi < 2; ++mi) {            // A: K chunks [8h..] and [16+8h..]
            int m = wr * 32 + mi * 16 + lrow;
            v8bf lo = *(const v8bf*)&sA[m * LDA + 8 * lhi];
            v8bf hi = *(const v8bf*)&sA[m * LDA + 16 + 8 * lhi];
            *(v8bf*)&af[mi] = lo; *((v8bf*)&af[mi] + 1) = hi;
        }
#pragma unroll
        for (int ni = 0; ni < 2; ++ni) {            // B: K chunks [8h..] and [16+8h..]
            int n = wc * 32 + ni * 16 + lrow;
            v8bf lo = *(const v8bf*)&sB[n * LDB + 8 * lhi];
            v8bf hi = *(const v8bf*)&sB[n * LDB + 16 + 8 * lhi];
            *(v8bf*)&bfr[ni] = lo; *((v8bf*)&bfr[ni] + 1) = hi;
        }
#pragma unroll
        for (int mi = 0; mi < 2; ++mi)
#pragma unroll
            for (int ni = 0; ni < 2; ++ni)
                acc[mi][ni] = __builtin_amdgcn_wmma_f32_16x16x32_bf16(
                    false, af[mi], false, bfr[ni], (short)0, acc[mi][ni], false, false);
        __syncthreads();
    }

    // epilogue: acc element e maps to M = base + e + 8*lhi, N = base + lrow
#pragma unroll
    for (int mi = 0; mi < 2; ++mi) {
#pragma unroll
        for (int ni = 0; ni < 2; ++ni) {
            const int gm = bM + wr * 32 + mi * 16 + 8 * lhi;
            const int gn = bN + wc * 32 + ni * 16 + lrow;
#pragma unroll
            for (int e = 0; e < 8; ++e) {
                const int r = gm + e;
                float v = acc[mi][ni][e];
                if (MODE == 0) {
                    outBf[(size_t)r * N + gn] = (bf16_t)v;
                } else if (MODE == 1) {
                    outF[(size_t)r * N + gn] = v;
                } else if (MODE == 2) {
                    v += bias[gn];
                    v = (v > 20.0f) ? v : log1pf(__expf(v));
                    v += __shfl_xor(v, 1); v += __shfl_xor(v, 2);
                    v += __shfl_xor(v, 4); v += __shfl_xor(v, 8);
                    if (lrow == 0) atomicAdd(&dsum[r], v);
                } else { // MODE == 3
                    outF[(size_t)r * N + gn] = v + resid[(size_t)r * N + gn];
                }
            }
        }
    }
}

// ---------------- depthwise causal conv (K=4) + SiLU ----------------
__global__ void conv_silu_kernel(const bf16_t* __restrict__ xr,
                                 const float* __restrict__ cw,
                                 const float* __restrict__ cb,
                                 bf16_t* __restrict__ xc) {
    int idx = blockIdx.x * blockDim.x + threadIdx.x;
    if (idx >= ROWS * D_INNER) return;
    const int c   = idx & (D_INNER - 1);
    const int row = idx >> 11;            // D_INNER == 2048
    const int t   = row & (TLEN - 1);
    float s = cb[c];
#pragma unroll
    for (int k = 0; k < D_CONV; ++k) {
        int tt = t + k - (D_CONV - 1);
        if (tt >= 0)
            s += cw[c * D_CONV + k] *
                 (float)xr[(size_t)(row - (t - tt)) * (2 * D_INNER) + c];
    }
    s = s * sigm(s);
    xc[idx] = (bf16_t)s;
}

// ---------------- sequential selective scan (tiny: 4 batches x 64 states) ---
__global__ __launch_bounds__(128)
void scan_kernel(const float* __restrict__ dsum, const float* __restrict__ bc,
                 const float* __restrict__ A_log, float* __restrict__ ys) {
    const int b = threadIdx.x >> 5;   // one wave per batch
    const int lane = threadIdx.x & 31;
    const float A0 = -__expf(A_log[lane]);
    const float A1 = -__expf(A_log[lane + 32]);
    float h0 = 0.f, h1 = 0.f;
    for (int t = 0; t < TLEN; ++t) {
        const size_t row = (size_t)b * TLEN + t;
        const float dm = dsum[row] * (1.0f / D_INNER);
        const float e0 = __expf(dm * A0), e1 = __expf(dm * A1);
        const float B0 = bc[row * 128 + lane];
        const float B1 = bc[row * 128 + lane + 32];
        const float C0 = bc[row * 128 + 64 + lane];
        const float C1 = bc[row * 128 + 96 + lane];
        h0 = h0 * e0 + B0;
        h1 = h1 * e1 + B1;
        float p = h0 * C0 + h1 * C1;
        p += __shfl_xor(p, 1); p += __shfl_xor(p, 2); p += __shfl_xor(p, 4);
        p += __shfl_xor(p, 8); p += __shfl_xor(p, 16);
        if (lane == 0) ys[row] = p;
    }
}

// ---------------- gating: y = (ys + D*xc) * silu(res) ----------------
__global__ void gate_kernel(const bf16_t* __restrict__ xc,
                            const bf16_t* __restrict__ xr,
                            const float* __restrict__ ys,
                            const float* __restrict__ Dp,
                            bf16_t* __restrict__ y) {
    int idx = blockIdx.x * blockDim.x + threadIdx.x;
    if (idx >= ROWS * D_INNER) return;
    const int c = idx & (D_INNER - 1);
    const int row = idx >> 11;
    const float xcv  = (float)xc[idx];
    const float resv = (float)xr[(size_t)row * (2 * D_INNER) + D_INNER + c];
    const float v = (ys[row] + Dp[c] * xcv) * (resv * sigm(resv));
    y[idx] = (bf16_t)v;
}

// ---------------- launcher ----------------
extern "C" void kernel_launch(void* const* d_in, const int* in_sizes, int n_in,
                              void* d_out, int out_size, void* d_ws, size_t ws_size,
                              hipStream_t stream) {
    (void)in_sizes; (void)n_in; (void)out_size; (void)ws_size;
    const float* x      = (const float*)d_in[0];
    const float* ln_g   = (const float*)d_in[1];
    const float* ln_b   = (const float*)d_in[2];
    const float* W_in   = (const float*)d_in[3];
    const float* conv_w = (const float*)d_in[4];
    const float* conv_b = (const float*)d_in[5];
    const float* W_x    = (const float*)d_in[6];
    const float* W_dt   = (const float*)d_in[7];
    const float* b_dt   = (const float*)d_in[8];
    const float* A_log  = (const float*)d_in[9];
    const float* Dp     = (const float*)d_in[10];
    const float* W_out  = (const float*)d_in[11];
    float* out = (float*)d_out;

    char* wsc = (char*)d_ws;
    size_t off = 0;
    auto alloc = [&](size_t bytes) { size_t o = off; off += (bytes + 255) & ~(size_t)255; return o; };
    const size_t o_xn   = alloc((size_t)ROWS * D_MODEL * 2);
    const size_t o_Win  = alloc((size_t)D_MODEL * 2 * D_INNER * 2);   // W_in^T
    const size_t o_Wx   = alloc((size_t)D_INNER * 2 * D_STATE * 2);   // W_x^T
    const size_t o_Wdt  = alloc((size_t)D_INNER * D_INNER * 2);       // W_dt^T
    const size_t o_Wout = alloc((size_t)D_INNER * D_MODEL * 2);       // W_out^T
    const size_t o_xr   = alloc((size_t)ROWS * 2 * D_INNER * 2);
    const size_t o_xc   = alloc((size_t)ROWS * D_INNER * 2);
    const size_t o_bc   = alloc((size_t)ROWS * 2 * D_STATE * 4);
    const size_t o_dsum = alloc((size_t)ROWS * 4);
    const size_t o_ys   = alloc((size_t)ROWS * 4);
    const size_t o_y    = alloc((size_t)ROWS * D_INNER * 2);

    bf16_t* xn_b   = (bf16_t*)(wsc + o_xn);
    bf16_t* Wint   = (bf16_t*)(wsc + o_Win);
    bf16_t* Wxt    = (bf16_t*)(wsc + o_Wx);
    bf16_t* Wdtt   = (bf16_t*)(wsc + o_Wdt);
    bf16_t* Woutt  = (bf16_t*)(wsc + o_Wout);
    bf16_t* xr_b   = (bf16_t*)(wsc + o_xr);
    bf16_t* xc_b   = (bf16_t*)(wsc + o_xc);
    float*  bc_f   = (float*)(wsc + o_bc);
    float*  dsum_f = (float*)(wsc + o_dsum);
    float*  ys_f   = (float*)(wsc + o_ys);
    bf16_t* y_b    = (bf16_t*)(wsc + o_y);

    // 0) zero the atomic row-sum accumulator (fresh every launch)
    zero_f32_kernel<<<(ROWS + 255) / 256, 256, 0, stream>>>(dsum_f, ROWS);

    // 1) LayerNorm -> bf16
    ln_kernel<<<ROWS, 256, 0, stream>>>(x, ln_g, ln_b, xn_b);

    // 2) weights -> bf16, transposed to [N x K] so GEMM staging is contiguous
    cvt_transpose_kernel<<<(D_MODEL * 2 * D_INNER + 255) / 256, 256, 0, stream>>>(W_in, Wint, D_MODEL, 2 * D_INNER);
    cvt_transpose_kernel<<<(D_INNER * 2 * D_STATE + 255) / 256, 256, 0, stream>>>(W_x, Wxt, D_INNER, 2 * D_STATE);
    cvt_transpose_kernel<<<(D_INNER * D_INNER + 255) / 256, 256, 0, stream>>>(W_dt, Wdtt, D_INNER, D_INNER);
    cvt_transpose_kernel<<<(D_INNER * D_MODEL + 255) / 256, 256, 0, stream>>>(W_out, Woutt, D_INNER, D_MODEL);

    // 3) xr = xn @ W_in   [8192 x 1024] x [1024 x 4096] -> bf16
    gemm_bf16_kernel<0><<<dim3((2 * D_INNER) / 128, ROWS / 64), 256, 0, stream>>>(
        xn_b, Wint, ROWS, 2 * D_INNER, D_MODEL,
        xr_b, nullptr, nullptr, nullptr, nullptr);

    // 4) depthwise conv + SiLU -> xc (bf16)
    conv_silu_kernel<<<(ROWS * D_INNER + 255) / 256, 256, 0, stream>>>(xr_b, conv_w, conv_b, xc_b);

    // 5) bc = xc @ W_x    [8192 x 2048] x [2048 x 128] -> f32
    gemm_bf16_kernel<1><<<dim3((2 * D_STATE) / 128, ROWS / 64), 256, 0, stream>>>(
        xc_b, Wxt, ROWS, 2 * D_STATE, D_INNER,
        nullptr, bc_f, nullptr, nullptr, nullptr);

    // 6) dsum[row] = sum_n softplus((xc @ W_dt)[row,n] + b_dt[n])  (fused, never stored)
    gemm_bf16_kernel<2><<<dim3(D_INNER / 128, ROWS / 64), 256, 0, stream>>>(
        xc_b, Wdtt, ROWS, D_INNER, D_INNER,
        nullptr, nullptr, b_dt, nullptr, dsum_f);

    // 7) sequential scan (tiny)
    scan_kernel<<<1, 128, 0, stream>>>(dsum_f, bc_f, A_log, ys_f);

    // 8) gating -> y (bf16)
    gate_kernel<<<(ROWS * D_INNER + 255) / 256, 256, 0, stream>>>(xc_b, xr_b, ys_f, Dp, y_b);

    // 9) out = y @ W_out + x   [8192 x 2048] x [2048 x 1024] -> f32, fused residual
    gemm_bf16_kernel<3><<<dim3(D_MODEL / 128, ROWS / 64), 256, 0, stream>>>(
        y_b, Woutt, ROWS, D_MODEL, D_INNER,
        nullptr, out, nullptr, x, nullptr);
}
